// MolecularE3nnTransformer_7164005449942
// MI455X (gfx1250) — matrix operand order repaired and checked by reading, not compiled, not run on hardware
//
#include <hip/hip_runtime.h>
#include <hip/hip_bf16.h>

// ---------------- problem constants (match reference) ----------------
#define N_NODES 4096
#define N_EDGES 16384
#define N_GRAPH 64
#define HD      64
#define NLAYER  2
#define NOL     2
#define NBASIS  10
#define MAXR    2.0f

typedef _Float16 h16v __attribute__((ext_vector_type(16)));
typedef _Float16 h8v  __attribute__((ext_vector_type(8)));
typedef float    f8v  __attribute__((ext_vector_type(8)));
typedef int      v4i_g __attribute__((vector_size(16)));  // matches async-builtin pointee

// LDS strides (halfs / floats), padded to break bank conflicts
#define FE_S 72
#define B_S  72
#define C_S  68

// ---------------- CDNA5 async global<->LDS paths (guarded) ----------------
#if defined(__has_builtin)
#if __has_builtin(__builtin_amdgcn_global_load_async_to_lds_b128)
#define HAVE_ASYNC_LOAD 1
#endif
#if __has_builtin(__builtin_amdgcn_global_store_async_from_lds_b128)
#define HAVE_ASYNC_STORE 1
#endif
#if __has_builtin(__builtin_amdgcn_s_wait_asynccnt)
#define HAVE_ASYNC_WAIT 1
#endif
#endif

// 16-byte global -> LDS copy; async (ASYNCcnt) when available
__device__ __forceinline__ void ldsCopy16(void* ldst, void* gsrc) {
#if defined(HAVE_ASYNC_LOAD)
  __builtin_amdgcn_global_load_async_to_lds_b128(
      (__attribute__((address_space(1))) v4i_g*)gsrc,
      (__attribute__((address_space(3))) v4i_g*)ldst, 0, 0);
#else
  *(h8v*)ldst = *(const h8v*)gsrc;
#endif
}

// 16-byte LDS -> global store; async when available
__device__ __forceinline__ void gstore16(void* gdst, void* lsrc) {
#if defined(HAVE_ASYNC_STORE)
  __builtin_amdgcn_global_store_async_from_lds_b128(
      (__attribute__((address_space(1))) v4i_g*)gdst,
      (__attribute__((address_space(3))) v4i_g*)lsrc, 0, 0);
#else
  *(float4*)gdst = *(const float4*)lsrc;
#endif
}

__device__ __forceinline__ void asyncWait0() {
#if defined(HAVE_ASYNC_LOAD) || defined(HAVE_ASYNC_STORE)
#if defined(HAVE_ASYNC_WAIT)
  __builtin_amdgcn_s_wait_asynccnt(0);
#else
  asm volatile("s_wait_asynccnt 0" ::: "memory");
#endif
#endif
}

// ---------------- WMMA fragment helpers ----------------
// A-matrix 16x32 f16 layout: lane half = lane/16, row m = lane%16.
// v16h elems 0..7  <- K[kb + 8*half .. +8)
// v16h elems 8..15 <- K[kb + 16 + 8*half .. +8)
__device__ __forceinline__ h16v ldA(const _Float16* __restrict__ row, int kb, int half) {
  h8v c1 = *(const h8v*)(row + kb + 8 * half);
  h8v c2 = *(const h8v*)(row + kb + 16 + 8 * half);
  h16v a;
#pragma unroll
  for (int i = 0; i < 8; ++i) { a[i] = c1[i]; a[i + 8] = c2[i]; }
  return a;
}

// Same, but A = s * fe  (rank-1 outer-product row, scalar s = h[m,j])
__device__ __forceinline__ h16v ldA_scaled(const _Float16* __restrict__ row, int kb, int half,
                                           _Float16 s) {
  h8v c1 = *(const h8v*)(row + kb + 8 * half);
  h8v c2 = *(const h8v*)(row + kb + 16 + 8 * half);
  h16v a;
#pragma unroll
  for (int i = 0; i < 8; ++i) {
    a[i]     = (_Float16)(c1[i] * s);
    a[i + 8] = (_Float16)(c2[i] * s);
  }
  return a;
}

// B-matrix 32x16 f16: lane col n = lane%16, elems 0..7 <- K[kb + 8*half ..+8), 8..15 <- K[kb + 16 + 8*half ..+8)
__device__ __forceinline__ h16v ldB(const _Float16* __restrict__ rowN, int kb, int half) {
  h8v c1 = *(const h8v*)(rowN + kb + 8 * half);
  h8v c2 = *(const h8v*)(rowN + kb + 16 + 8 * half);
  h16v b;
#pragma unroll
  for (int i = 0; i < 8; ++i) { b[i] = c1[i]; b[i + 8] = c2[i]; }
  return b;
}

__device__ __forceinline__ f8v wmma_f16(h16v a, h16v b, f8v c) {
  return __builtin_amdgcn_wmma_f32_16x16x32_f16(false, a, false, b, (short)0, c, false, false);
}

// ---------------- weight conversion kernels ----------------
// batched 64x64 transpose f32 -> f16 (B needs N-major rows of K)
__global__ void convT_kernel(const float* __restrict__ src, _Float16* __restrict__ dstT,
                             int batches) {
  int i = blockIdx.x * blockDim.x + threadIdx.x;
  if (i >= batches * HD * HD) return;
  int b = i / (HD * HD), r = i % (HD * HD);
  int n = r >> 6, k = r & 63;
  dstT[b * HD * HD + n * HD + k] = (_Float16)src[b * HD * HD + k * HD + n];
}

// k2 [L][16][u*64+w] -> Bt [L][w(n)][j*64+u(K)]  (N-major, K contiguous)
__global__ void convK2_kernel(const float* __restrict__ src, _Float16* __restrict__ dst) {
  int i = blockIdx.x * blockDim.x + threadIdx.x;
  if (i >= NLAYER * 64 * 1024) return;
  int l = i >> 16, r = i & 65535;
  int n = r >> 10, K = r & 1023;
  int j = K >> 6, u = K & 63;
  dst[i] = (_Float16)src[l * 65536 + j * 4096 + u * 64 + n];
}

// ---------------- node/edge setup ----------------
__global__ void init_kernel(const int* __restrict__ x, const float* __restrict__ embd,
                            float* __restrict__ f32, _Float16* __restrict__ f16) {
  int i = blockIdx.x * blockDim.x + threadIdx.x;  // N*HD
  int n = i >> 6, h = i & 63;
  float v = embd[x[n] * HD + h];
  f32[i] = v;
  f16[i] = (_Float16)v;
}

__device__ __forceinline__ float soft_unit(float t) {
  return t > 0.f ? __expf(-1.f / t) : 0.f;
}

__global__ void basis_kernel(const float* __restrict__ ea, float* __restrict__ scal,
                             float* __restrict__ ewc) {
  int e = blockIdx.x * blockDim.x + threadIdx.x;
  if (e >= N_EDGES) return;
  float x0 = ea[e * 3 + 0], x1 = ea[e * 3 + 1], x2 = ea[e * 3 + 2];
  float elen = sqrtf(x0 * x0 + x1 * x1 + x2 * x2);
  const float step = MAXR / (NBASIS + 1);
  const float C = 1.14136f * 7.38905609893065f * 3.1622776601683795f;  // 1.14136*e^2*sqrt(NB)
#pragma unroll
  for (int b = 0; b < NBASIS; ++b) {
    float center = step * (float)(b + 1);
    float diff = (elen - center) / step;
    scal[e * NBASIS + b] = C * soft_unit(diff + 1.f) * soft_unit(1.f - diff);
  }
  ewc[e] = soft_unit(10.f * (1.f - elen / MAXR));
}

// fcnet hidden: h = silu(scal @ W1 / sqrt(NB)); fold the 1/(sqrt(16)*sqrt(H)) = 1/32 path norm in
__global__ void fcnet_kernel(const float* __restrict__ scal, const float* __restrict__ k1,
                             const float* __restrict__ v1, _Float16* __restrict__ hK,
                             _Float16* __restrict__ hV) {
  int e = blockIdx.x * blockDim.x + threadIdx.x;
  if (e >= N_EDGES) return;
  float s[NBASIS];
#pragma unroll
  for (int b = 0; b < NBASIS; ++b) s[b] = scal[e * NBASIS + b];
#pragma unroll
  for (int i = 0; i < 16; ++i) {
    float ak = 0.f, av = 0.f;
#pragma unroll
    for (int b = 0; b < NBASIS; ++b) {
      ak += s[b] * k1[b * 16 + i];
      av += s[b] * v1[b * 16 + i];
    }
    ak *= 0.31622776601683794f;  // 1/sqrt(10)
    av *= 0.31622776601683794f;
    float sk = ak / (1.f + __expf(-ak));  // silu
    float sv = av / (1.f + __expf(-av));
    hK[e * 16 + i] = (_Float16)(sk * (1.f / 32.f));
    hV[e * 16 + i] = (_Float16)(sv * (1.f / 32.f));
  }
}

// ---------------- node GEMM: qd = ((f @ Wq)/8) @ Wd / 64 ----------------
__global__ void __launch_bounds__(512)
node_qd_kernel(const _Float16* __restrict__ fF16, const _Float16* __restrict__ WqT,
               const _Float16* __restrict__ WdT, float* __restrict__ qd32) {
  __shared__ __align__(16) _Float16 sA[64 * FE_S];
  __shared__ __align__(16) _Float16 sB[64 * B_S];
  __shared__ __align__(16) _Float16 sQ[64 * FE_S];

  const int tid = threadIdx.x;
  const int rowBase = blockIdx.x * 64;
  {
    int m = tid >> 3, seg = tid & 7;
    ldsCopy16(sA + m * FE_S + seg * 8, (void*)(fF16 + (rowBase + m) * HD + seg * 8));
    ldsCopy16(sB + m * B_S + seg * 8, (void*)(WqT + m * HD + seg * 8));
  }
  asyncWait0();
  __syncthreads();

  const int wave = tid >> 5, lane = tid & 31;
  const int mi = wave >> 2, ni = wave & 3;
  const int half = lane >> 4, lm = lane & 15;
  const int mRow = mi * 16 + lm, nCol = ni * 16 + lm;

  f8v acc = {};
#pragma unroll
  for (int kb = 0; kb < 64; kb += 32) {
    h16v a = ldA(sA + mRow * FE_S, kb, half);
    h16v b = ldB(sB + nCol * B_S, kb, half);
    acc = wmma_f16(a, b, acc);
  }
  // q (scaled 1/sqrt(H)) -> LDS f16 for second GEMM
#pragma unroll
  for (int r = 0; r < 8; ++r) {
    int ml = r + 8 * half;
    sQ[(mi * 16 + ml) * FE_S + nCol] = (_Float16)(acc[r] * 0.125f);
  }
  __syncthreads();
  {
    int m = tid >> 3, seg = tid & 7;
    ldsCopy16(sB + m * B_S + seg * 8, (void*)(WdT + m * HD + seg * 8));
  }
  asyncWait0();
  __syncthreads();

  f8v acc2 = {};
#pragma unroll
  for (int kb = 0; kb < 64; kb += 32) {
    h16v a = ldA(sQ + mRow * FE_S, kb, half);
    h16v b = ldB(sB + nCol * B_S, kb, half);
    acc2 = wmma_f16(a, b, acc2);
  }
#pragma unroll
  for (int r = 0; r < 8; ++r) {
    int ml = r + 8 * half;
    qd32[(rowBase + mi * 16 + ml) * HD + nCol] = acc2[r] * (1.f / 64.f);  // fold logit 1/H
  }
}

// ---------------- fused edge kernel ----------------
// 64 edges / block; k = (h_k ⊗ fe) @ Bk, v = (h_v ⊗ fe) @ Bv  via WMMA with
// double-buffered async B staging; then logit = qd[dst]·k, ex = ewc*exp(logit),
// z[dst] += ex, v -> global (async from LDS).
__global__ void __launch_bounds__(512)
edge_attn_kernel(const _Float16* __restrict__ fF16, const float* __restrict__ qd32,
                 const _Float16* __restrict__ hK, const _Float16* __restrict__ hV,
                 const _Float16* __restrict__ Bk, const _Float16* __restrict__ Bv,
                 const int* __restrict__ src, const int* __restrict__ dst,
                 const float* __restrict__ ewc, float* __restrict__ vOut,
                 float* __restrict__ exG, float* __restrict__ z) {
  __shared__ __align__(16) _Float16 sFe[64 * FE_S];
  __shared__ __align__(16) _Float16 sHk[64 * 16];
  __shared__ __align__(16) _Float16 sHv[64 * 16];
  __shared__ __align__(16) _Float16 sBk[2][64 * B_S];
  __shared__ __align__(16) _Float16 sBv[2][64 * B_S];
  __shared__ __align__(16) float    sK[64 * C_S];
  __shared__ __align__(16) float    sV[64 * C_S];

  const int tid = threadIdx.x;
  const int eBase = blockIdx.x * 64;
  const int nLd = tid >> 3, segLd = tid & 7;

  // gather fe = f16[src] tile + h tiles + first B block, all async
  {
    int s = src[eBase + nLd];
    ldsCopy16(sFe + nLd * FE_S + segLd * 8, (void*)(fF16 + s * HD + segLd * 8));
  }
  if (tid < 128) {
    int m = tid >> 1, seg = tid & 1;
    ldsCopy16(sHk + m * 16 + seg * 8, (void*)(hK + (eBase + m) * 16 + seg * 8));
    ldsCopy16(sHv + m * 16 + seg * 8, (void*)(hV + (eBase + m) * 16 + seg * 8));
  }
  ldsCopy16(&sBk[0][nLd * B_S + segLd * 8], (void*)(Bk + nLd * 1024 + segLd * 8));
  ldsCopy16(&sBv[0][nLd * B_S + segLd * 8], (void*)(Bv + nLd * 1024 + segLd * 8));
  asyncWait0();
  __syncthreads();

  const int wave = tid >> 5, lane = tid & 31;
  const int mi = wave >> 2, ni = wave & 3;
  const int half = lane >> 4, lm = lane & 15;
  const int mRow = mi * 16 + lm, nCol = ni * 16 + lm;

  f8v accK = {};
  f8v accV = {};

  for (int j = 0; j < 16; ++j) {   // K blocks of 64 (one basis channel j each)
    const int buf = j & 1;
    if (j < 15) {  // async stage next block while WMMAs run on current one
      ldsCopy16(&sBk[buf ^ 1][nLd * B_S + segLd * 8],
                (void*)(Bk + nLd * 1024 + (j + 1) * 64 + segLd * 8));
      ldsCopy16(&sBv[buf ^ 1][nLd * B_S + segLd * 8],
                (void*)(Bv + nLd * 1024 + (j + 1) * 64 + segLd * 8));
    }
    const _Float16 hk = sHk[mRow * 16 + j];
    const _Float16 hv = sHv[mRow * 16 + j];
    const _Float16* feRow = sFe + mRow * FE_S;
    const _Float16* bkRow = &sBk[buf][nCol * B_S];
    const _Float16* bvRow = &sBv[buf][nCol * B_S];
#pragma unroll
    for (int kb = 0; kb < 64; kb += 32) {
      h16v aK = ldA_scaled(feRow, kb, half, hk);
      h16v bK = ldB(bkRow, kb, half);
      accK = wmma_f16(aK, bK, accK);
      h16v aV = ldA_scaled(feRow, kb, half, hv);
      h16v bV = ldB(bvRow, kb, half);
      accV = wmma_f16(aV, bV, accV);
    }
    asyncWait0();
    __syncthreads();
  }

  // C tiles -> LDS
#pragma unroll
  for (int r = 0; r < 8; ++r) {
    int ml = r + 8 * half;
    sK[(mi * 16 + ml) * C_S + nCol] = accK[r];
    sV[(mi * 16 + ml) * C_S + nCol] = accV[r];
  }
  __syncthreads();

  // logit / exp / segment-sum  (k consumed in-LDS; never hits HBM)
  if (tid < 64) {
    int e = eBase + tid;
    int d = dst[e];
    const float* qrow = qd32 + d * HD;  // already scaled by 1/H
    const float* krow = sK + tid * C_S;
    float acc = 0.f;
#pragma unroll
    for (int c = 0; c < 64; c += 4) {
      float4 qv = *(const float4*)(qrow + c);
      acc += qv.x * krow[c] + qv.y * krow[c + 1] + qv.z * krow[c + 2] + qv.w * krow[c + 3];
    }
    float ex = ewc[e] * __expf(acc);
    exG[e] = ex;
    atomicAdd(&z[d], ex);
  }
  // v tile LDS -> global (async store from LDS; ENDPGM waits all counters)
  {
    int m = tid >> 3, seg = tid & 7;
    int e = eBase + m;
    gstore16(vOut + e * HD + seg * 8, sV + m * C_S + seg * 8);
    gstore16(vOut + e * HD + seg * 8 + 4, sV + m * C_S + seg * 8 + 4);
  }
}

// ---------------- scatter-softmax aggregation ----------------
__global__ void softagg_kernel(const float* __restrict__ ex, const float* __restrict__ z,
                               const int* __restrict__ dst, const float* __restrict__ vOut,
                               float* __restrict__ agg) {
  int t = blockIdx.x * blockDim.x + threadIdx.x;  // E*8
  int e = t >> 3, seg = t & 7;
  int d = dst[e];
  float zd = z[d];
  zd = (zd == 0.f) ? 1.f : zd;
  float w = sqrtf(fmaxf(ex[e] / zd, 0.f));  // sqrt(relu(alpha))
  const float* vr = vOut + e * HD + seg * 8;
  float* ar = agg + d * HD + seg * 8;
#pragma unroll
  for (int c = 0; c < 8; ++c) atomicAdd(ar + c, w * vr[c]);
}

// ---------------- node normalize + residual ----------------
__global__ void nodeupd_kernel(const float* __restrict__ agg, float* __restrict__ f32,
                               _Float16* __restrict__ f16) {
  int gt = blockIdx.x * blockDim.x + threadIdx.x;  // N*32, one wave32 per node
  int n = gt >> 5, lane = gt & 31;
  float a0 = agg[n * HD + lane], a1 = agg[n * HD + 32 + lane];
  float ss = a0 * a0 + a1 * a1;
#pragma unroll
  for (int off = 16; off > 0; off >>= 1) ss += __shfl_xor(ss, off, 32);
  float nrm = sqrtf(fmaxf(ss, 1e-24f));
  float h0 = fmaxf(a0 / nrm, 0.f), h1 = fmaxf(a1 / nrm, 0.f);
  float v0 = f32[n * HD + lane] + h0;
  float v1 = f32[n * HD + 32 + lane] + h1;
  f32[n * HD + lane] = v0;
  f32[n * HD + 32 + lane] = v1;
  f16[n * HD + lane] = (_Float16)v0;
  f16[n * HD + 32 + lane] = (_Float16)v1;
}

// ---------------- output MLP layer: f = relu(f @ W + b) ----------------
__global__ void __launch_bounds__(512)
node_linear_kernel(const _Float16* __restrict__ fIn16, const _Float16* __restrict__ WT,
                   const float* __restrict__ bias, float* __restrict__ fOut32,
                   _Float16* __restrict__ fOut16) {
  __shared__ __align__(16) _Float16 sA[64 * FE_S];
  __shared__ __align__(16) _Float16 sB[64 * B_S];
  const int tid = threadIdx.x;
  const int rowBase = blockIdx.x * 64;
  {
    int m = tid >> 3, seg = tid & 7;
    ldsCopy16(sA + m * FE_S + seg * 8, (void*)(fIn16 + (rowBase + m) * HD + seg * 8));
    ldsCopy16(sB + m * B_S + seg * 8, (void*)(WT + m * HD + seg * 8));
  }
  asyncWait0();
  __syncthreads();

  const int wave = tid >> 5, lane = tid & 31;
  const int mi = wave >> 2, ni = wave & 3;
  const int half = lane >> 4, lm = lane & 15;
  const int mRow = mi * 16 + lm, nCol = ni * 16 + lm;

  f8v acc = {};
#pragma unroll
  for (int kb = 0; kb < 64; kb += 32) {
    h16v a = ldA(sA + mRow * FE_S, kb, half);
    h16v b = ldB(sB + nCol * B_S, kb, half);
    acc = wmma_f16(a, b, acc);
  }
  float bn = bias[nCol];
#pragma unroll
  for (int r = 0; r < 8; ++r) {
    int row = rowBase + mi * 16 + r + 8 * half;
    float v = fmaxf(acc[r] + bn, 0.f);
    fOut32[row * HD + nCol] = v;
    fOut16[row * HD + nCol] = (_Float16)v;
  }
}

// ---------------- pooling / output ----------------
__global__ void zero_kernel(float* __restrict__ p, int n) {
  int i = blockIdx.x * blockDim.x + threadIdx.x;
  if (i < n) p[i] = 0.f;
}

__global__ void pool_kernel(const float* __restrict__ f, const int* __restrict__ batch,
                            float* __restrict__ pooled) {
  int i = blockIdx.x * blockDim.x + threadIdx.x;  // N*HD
  int n = i >> 6, h = i & 63;
  atomicAdd(&pooled[batch[n] * HD + h], f[i]);
}

__global__ void final_kernel(const float* __restrict__ pooled, const float* __restrict__ Wp,
                             const float* __restrict__ bp, float* __restrict__ out) {
  int g = threadIdx.x;
  if (g < N_GRAPH) {
    float acc = 0.f;
#pragma unroll 8
    for (int h = 0; h < HD; ++h) acc += pooled[g * HD + h] * Wp[h];
    out[g] = acc + bp[0];
  }
}

// ---------------- launch ----------------
extern "C" void kernel_launch(void* const* d_in, const int* in_sizes, int n_in,
                              void* d_out, int out_size, void* d_ws, size_t ws_size,
                              hipStream_t stream) {
  (void)in_sizes; (void)n_in; (void)out_size; (void)ws_size;
  const int*   x     = (const int*)d_in[0];
  const int*   eidx  = (const int*)d_in[1];
  const float* eattr = (const float*)d_in[2];
  const int*   batch = (const int*)d_in[3];
  const float* embd  = (const float*)d_in[4];
  const float* Wq    = (const float*)d_in[5];
  const float* k1    = (const float*)d_in[6];
  const float* k2    = (const float*)d_in[7];
  const float* v1    = (const float*)d_in[8];
  const float* v2    = (const float*)d_in[9];
  const float* Wd    = (const float*)d_in[10];
  const float* linW  = (const float*)d_in[11];
  const float* linb  = (const float*)d_in[12];
  const float* Wp    = (const float*)d_in[13];
  const float* bp    = (const float*)d_in[14];
  const int* src = eidx;
  const int* dst = eidx + N_EDGES;
  float* out = (float*)d_out;

  char* ws = (char*)d_ws;
  size_t off = 0;
  auto alloc = [&](size_t bytes) -> void* {
    void* p = (void*)(ws + off);
    off += (bytes + 255) & ~((size_t)255);
    return p;
  };
  float*    fF32  = (float*)alloc((size_t)N_NODES * HD * 4);
  _Float16* fF16  = (_Float16*)alloc((size_t)N_NODES * HD * 2);
  float*    qd32  = (float*)alloc((size_t)N_NODES * HD * 4);
  float*    scal  = (float*)alloc((size_t)N_EDGES * NBASIS * 4);
  float*    ewc   = (float*)alloc((size_t)N_EDGES * 4);
  _Float16* hK    = (_Float16*)alloc((size_t)N_EDGES * 16 * 2);
  _Float16* hV    = (_Float16*)alloc((size_t)N_EDGES * 16 * 2);
  float*    vOut  = (float*)alloc((size_t)N_EDGES * HD * 4);
  float*    exG   = (float*)alloc((size_t)N_EDGES * 4);
  float*    z     = (float*)alloc((size_t)N_NODES * 4);
  float*    agg   = (float*)alloc((size_t)N_NODES * HD * 4);
  float*    pooled= (float*)alloc((size_t)N_GRAPH * HD * 4);
  _Float16* WqT   = (_Float16*)alloc((size_t)NLAYER * HD * HD * 2);
  _Float16* WdT   = (_Float16*)alloc((size_t)NLAYER * HD * HD * 2);
  _Float16* linWT = (_Float16*)alloc((size_t)NOL * HD * HD * 2);
  _Float16* Bk    = (_Float16*)alloc((size_t)NLAYER * 64 * 1024 * 2);
  _Float16* Bv    = (_Float16*)alloc((size_t)NLAYER * 64 * 1024 * 2);

  // weight conversion (tiny, L2 resident afterwards)
  convT_kernel<<<(NLAYER * HD * HD + 255) / 256, 256, 0, stream>>>(Wq, WqT, NLAYER);
  convT_kernel<<<(NLAYER * HD * HD + 255) / 256, 256, 0, stream>>>(Wd, WdT, NLAYER);
  convT_kernel<<<(NOL * HD * HD + 255) / 256, 256, 0, stream>>>(linW, linWT, NOL);
  convK2_kernel<<<(NLAYER * 64 * 1024 + 255) / 256, 256, 0, stream>>>(k2, Bk);
  convK2_kernel<<<(NLAYER * 64 * 1024 + 255) / 256, 256, 0, stream>>>(v2, Bv);

  init_kernel<<<(N_NODES * HD) / 256, 256, 0, stream>>>(x, embd, fF32, fF16);
  basis_kernel<<<N_EDGES / 256, 256, 0, stream>>>(eattr, scal, ewc);

  for (int l = 0; l < NLAYER; ++l) {
    fcnet_kernel<<<N_EDGES / 256, 256, 0, stream>>>(scal, k1 + l * NBASIS * 16,
                                                    v1 + l * NBASIS * 16, hK, hV);
    zero_kernel<<<(N_NODES + 255) / 256, 256, 0, stream>>>(z, N_NODES);
    zero_kernel<<<(N_NODES * HD + 255) / 256, 256, 0, stream>>>(agg, N_NODES * HD);
    node_qd_kernel<<<N_NODES / 64, 512, 0, stream>>>(fF16, WqT + l * HD * HD,
                                                     WdT + l * HD * HD, qd32);
    edge_attn_kernel<<<N_EDGES / 64, 512, 0, stream>>>(fF16, qd32, hK, hV,
                                                       Bk + l * 64 * 1024, Bv + l * 64 * 1024,
                                                       src, dst, ewc, vOut, exG, z);
    softagg_kernel<<<(N_EDGES * 8) / 256, 256, 0, stream>>>(exG, z, dst, vOut, agg);
    nodeupd_kernel<<<(N_NODES * 32) / 256, 256, 0, stream>>>(agg, fF32, fF16);
  }

  for (int i = 0; i < NOL; ++i)
    node_linear_kernel<<<N_NODES / 64, 512, 0, stream>>>(fF16, linWT + i * HD * HD,
                                                         linb + i * HD, fF32, fF16);

  zero_kernel<<<(N_GRAPH * HD + 255) / 256, 256, 0, stream>>>(pooled, N_GRAPH * HD);
  pool_kernel<<<(N_NODES * HD) / 256, 256, 0, stream>>>(fF32, batch, pooled);
  final_kernel<<<1, 64, 0, stream>>>(pooled, Wp, bp, out);
}
